// HybridCNNLong_30640296690010
// MI455X (gfx1250) — matrix operand order repaired and checked by reading, not compiled, not run on hardware
//
#include <hip/hip_runtime.h>
#include <hip/hip_bf16.h>

// ---------------------------------------------------------------------------
// HybridCNNLong for MI455X (gfx1250): f16 WMMA GEMMs with f32 accumulation,
// Tensor Data Mover (TDM) staging of all f16 tiles into LDS.
//   B=1024, L=26, A=8192, C1=256, D=512, E=1024
// ---------------------------------------------------------------------------

typedef __attribute__((ext_vector_type(16))) _Float16 v16h;
typedef __attribute__((ext_vector_type(8)))  _Float16 v8h;
typedef __attribute__((ext_vector_type(8)))  float    v8f;
typedef __attribute__((ext_vector_type(4)))  unsigned v4u;
typedef __attribute__((ext_vector_type(8)))  int      v8i;
typedef __attribute__((ext_vector_type(4)))  int      v4i;

#define C_B   1024
#define C_L   26
#define C_A   8192
#define C_C1  256
#define C_D   512
#define C_E   1024
#define C_M   (C_B * C_L)   // 26624 rows for the per-position GEMMs

#if __has_builtin(__builtin_amdgcn_tensor_load_to_lds)
#define HAVE_TDM 1
#else
#define HAVE_TDM 0
#endif

__device__ __forceinline__ v8f wmma16(v16h a, v16h b, v8f c) {
    // v_wmma_f32_16x16x32_f16: D = A(16x32 f16) * B(32x16 f16) + C(16x16 f32)
    return __builtin_amdgcn_wmma_f32_16x16x32_f16(
        false, a, false, b, (short)0, c, false, false);
}

__device__ __forceinline__ float fast_sigmoid(float x) {
    return 1.0f / (1.0f + __expf(-x));
}
__device__ __forceinline__ float fast_tanh(float x) {
    return 1.0f - 2.0f / (__expf(2.0f * x) + 1.0f);
}

#if HAVE_TDM
// ---------------------------------------------------------------------------
// TDM: DMA a 2D tile of f16 (rows x 32 elements, row pitch stride_elems) from
// global memory into LDS.  Descriptor per CDNA5 ISA section 8 (D# groups).
// Issue from ONE wave only; complete with s_wait_tensorcnt 0 + barrier.
// ---------------------------------------------------------------------------
__device__ __forceinline__ void tdm_load_tile_f16(void* lds_dst,
                                                  const _Float16* gsrc,
                                                  int rows, int stride_elems) {
    // Generic LDS pointer: low 32 bits are the LDS byte offset (aperture rule).
    unsigned lds_off = (unsigned)(unsigned long long)lds_dst;
    unsigned long long ga = (unsigned long long)gsrc;

    v4u g0;
    g0[0] = 1u;                                    // count=1, user mode
    g0[1] = lds_off;                               // bits 63:32  lds_addr
    g0[2] = (unsigned)(ga & 0xFFFFFFFFull);        // bits 95:64  global_addr lo
    g0[3] = (unsigned)((ga >> 32) & 0x01FFFFFFull) // bits 120:96 global_addr hi
          | (2u << 30);                            // bits 127:126 type=2 (image)

    v8i g1;
    g1[0] = 1 << 16;                 // data_size = 1 (2 bytes); mask/flags = 0
    g1[1] = 32 << 16;                // tensor_dim0[15:0]  at bits 63:48  (=32)
    g1[2] = rows << 16;              // tensor_dim1[15:0]  at bits 95:80
    g1[3] = 32 << 16;                // tile_dim0          at bits 127:112 (=32)
    g1[4] = rows;                    // tile_dim1          at bits 143:128
    g1[5] = stride_elems;            // tensor_dim0_stride at bits 191:160
    g1[6] = 0;                       // stride hi / dim1_stride lo
    g1[7] = 0;

    v4i z = {0, 0, 0, 0};            // groups 2/3 unused (2D tensor)
#if __clang_major__ >= 23
    v8i z8 = {0, 0, 0, 0, 0, 0, 0, 0};
    __builtin_amdgcn_tensor_load_to_lds(g0, g1, z, z, z8, 0);
#else
    __builtin_amdgcn_tensor_load_to_lds(g0, g1, z, z, 0);
#endif
}
#endif // HAVE_TDM

// ---------------------------------------------------------------------------
// Helper kernels: weight f32->f16 conversion, h-state zeroing.
// ---------------------------------------------------------------------------
__global__ void cvt_f32_to_f16_kernel(const float* __restrict__ src,
                                      _Float16* __restrict__ dst, int n) {
    int i = blockIdx.x * 256 + threadIdx.x;
    if (i < n) dst[i] = (_Float16)src[i];
}

__global__ void zero_h_kernel(float* __restrict__ h32,
                              _Float16* __restrict__ h16, int n) {
    int i = blockIdx.x * 256 + threadIdx.x;
    if (i < n) { h32[i] = 0.0f; h16[i] = (_Float16)0.0f; }
}

// ---------------------------------------------------------------------------
// Templated WMMA GEMM:  C[M,N] = act(A[M,K] * Bw[N,K]^T + bias)
//   Block: 128 threads = 4 waves.  Block tile: 128(M) x 64(N).  K-chunk: 32.
//   Wave tile: 32(M) x 64(N) -> 8 accumulators; each B fragment reused by
//   2 WMMAs (hides ds latency; 8 wmma per 12 ds_load_b128).
//   A_F16: A is f16 (TDM-staged); else f32 converted while staging (conv1).
//   EPI==0: threshold(>1e-6) then store f16.  EPI==1: store f32.
// ---------------------------------------------------------------------------
template <bool A_F16, int EPI>
__global__ __launch_bounds__(128)
void gemm_tile_kernel(const void* __restrict__ Av, int lda,
                      const _Float16* __restrict__ Bw, int ldb,
                      const float* __restrict__ bias,
                      void* __restrict__ Cv, int ldc, int K) {
    __shared__ alignas(64) _Float16 As[128][32];
    __shared__ alignas(64) _Float16 Bt[64][32];   // Bt[n][k] = Bw[col0+n][k0+k]

    const int tid  = threadIdx.x;
    const int wave = tid >> 5;
    const int lane = tid & 31;
    const int m_   = lane & 15;   // row/col within 16-tile
    const int g    = lane >> 4;   // lane half-group
    const int row0 = blockIdx.x * 128;
    const int col0 = blockIdx.y * 64;

    v8f acc[2][4] = {};

    for (int k0 = 0; k0 < K; k0 += 32) {
#if HAVE_TDM
        if (wave == 0) {
            tdm_load_tile_f16(&Bt[0][0], &Bw[(long)col0 * ldb + k0], 64, ldb);
            if (A_F16)
                tdm_load_tile_f16(&As[0][0],
                                  (const _Float16*)Av + (long)row0 * lda + k0,
                                  128, lda);
        }
#else
        {   // manual staging fallback
            #pragma unroll
            for (int it = 0; it < 2; ++it) {
                int f = it * 128 + tid;
                int r = f >> 2, cq = f & 3;
                *(uint4*)&Bt[r][cq * 8] =
                    *(const uint4*)&Bw[(long)(col0 + r) * ldb + k0 + cq * 8];
            }
            if (A_F16) {
                const _Float16* A = (const _Float16*)Av;
                #pragma unroll
                for (int it = 0; it < 4; ++it) {
                    int f = it * 128 + tid;
                    int r = f >> 2, cq = f & 3;
                    *(uint4*)&As[r][cq * 8] =
                        *(const uint4*)&A[(long)(row0 + r) * lda + k0 + cq * 8];
                }
            }
        }
#endif
        if (!A_F16) {   // conv1: fp32 -> f16 convert while staging
            const float* A = (const float*)Av;
            #pragma unroll
            for (int it = 0; it < 8; ++it) {
                int f = it * 128 + tid;          // 0..1023 (float4)
                int r = f >> 3, cq = f & 7;
                const float* src = &A[(long)(row0 + r) * lda + k0 + cq * 4];
                float4 v = *(const float4*)src;
                if (k0 + 32 < K)                  // global_prefetch_b8
                    __builtin_prefetch(src + 32, 0, 1);
                As[r][cq * 4 + 0] = (_Float16)v.x;
                As[r][cq * 4 + 1] = (_Float16)v.y;
                As[r][cq * 4 + 2] = (_Float16)v.z;
                As[r][cq * 4 + 3] = (_Float16)v.w;
            }
        }
#if HAVE_TDM
        if (wave == 0) __builtin_amdgcn_s_wait_tensorcnt(0);
#endif
        __syncthreads();

        // ---- compute: 8 WMMAs per wave per chunk, B fragments reused 2x ----
        union Frag { v16h v; v8h h[2]; };
        Frag a0, a1;
        a0.h[0] = *(const v8h*)&As[wave * 32 + m_][g * 16];
        a0.h[1] = *(const v8h*)&As[wave * 32 + m_][g * 16 + 8];
        a1.h[0] = *(const v8h*)&As[wave * 32 + 16 + m_][g * 16];
        a1.h[1] = *(const v8h*)&As[wave * 32 + 16 + m_][g * 16 + 8];
        #pragma unroll
        for (int s = 0; s < 4; ++s) {
            v16h b = *(const v16h*)&Bt[s * 16 + m_][g * 16];
            acc[0][s] = wmma16(a0.v, b, acc[0][s]);
            acc[1][s] = wmma16(a1.v, b, acc[1][s]);
        }
        __syncthreads();
    }

    // ---- epilogue ----
    #pragma unroll
    for (int s = 0; s < 4; ++s) {
        int col = col0 + s * 16 + m_;
        float bv = bias[col];
        #pragma unroll
        for (int h = 0; h < 2; ++h) {
            #pragma unroll
            for (int r = 0; r < 8; ++r) {
                int row = row0 + wave * 32 + h * 16 + 8 * g + r;
                float val = acc[h][s][r] + bv;
                if (EPI == 0) {
                    val = (val > 1e-6f) ? val : 0.0f;
                    ((_Float16*)Cv)[(long)row * ldc + col] = (_Float16)val;
                } else {
                    ((float*)Cv)[(long)row * ldc + col] = val;
                }
            }
        }
    }
}

// ---------------------------------------------------------------------------
// Fused GRU step (one launch per timestep):
//   gi = x_t @ W_ih^T, gh = h @ W_hh^T  (WMMA, K=512)
//   r = sig(i_r+h_r); z = sig(i_z+h_z); n = tanh(i_n + r*h_n)
//   h_new = (1-z)*n + z*h    (h kept in f32; f16 mirror for next GEMM)
//   Block: 128 threads = 4 waves.  Tile: 64(batch) x 32(D).
//   All 8 operand tiles staged per K-chunk by the TDM from wave 0.
// ---------------------------------------------------------------------------
__global__ __launch_bounds__(128)
void gru_step_kernel(const _Float16* __restrict__ x16,   // [B, L, D] f16
                     int t,
                     const _Float16* __restrict__ hp16,  // [B, D] f16
                     const float* __restrict__ hp32,     // [B, D] f32
                     const _Float16* __restrict__ wih16, // [3D, D] f16
                     const _Float16* __restrict__ whh16, // [3D, D] f16
                     const float* __restrict__ b_ih,     // [3D]
                     const float* __restrict__ b_hh,     // [3D]
                     _Float16* __restrict__ hn16,
                     float* __restrict__ hn32) {
    __shared__ alignas(64) _Float16 Ax[64][32];
    __shared__ alignas(64) _Float16 Ah[64][32];
    __shared__ alignas(64) _Float16 Bt[6][32][32];  // [mat*3+gate][n][k]

    const int tid  = threadIdx.x;
    const int wave = tid >> 5;
    const int lane = tid & 31;
    const int m_   = lane & 15;
    const int g    = lane >> 4;
    const int row0 = blockIdx.x * 64;   // batch rows
    const int col0 = blockIdx.y * 32;   // D columns

    v8f accR[2] = {}, accZ[2] = {}, accNi[2] = {}, accNh[2] = {};

    for (int k0 = 0; k0 < C_D; k0 += 32) {
#if HAVE_TDM
        if (wave == 0) {
            tdm_load_tile_f16(&Ax[0][0],
                              &x16[((long)row0 * C_L + t) * C_D + k0],
                              64, C_L * C_D);
            tdm_load_tile_f16(&Ah[0][0],
                              &hp16[(long)row0 * C_D + k0], 64, C_D);
            #pragma unroll
            for (int j = 0; j < 6; ++j) {
                int gate = j % 3;
                const _Float16* W = (j < 3) ? wih16 : whh16;
                tdm_load_tile_f16(&Bt[j][0][0],
                                  &W[(long)(gate * C_D + col0) * C_D + k0],
                                  32, C_D);
            }
            __builtin_amdgcn_s_wait_tensorcnt(0);
        }
#else
        #pragma unroll
        for (int it = 0; it < 2; ++it) {
            int f = it * 128 + tid;
            int r = f >> 2, cq = f & 3;
            *(uint4*)&Ax[r][cq * 8] = *(const uint4*)
                &x16[((long)(row0 + r) * C_L + t) * C_D + k0 + cq * 8];
            *(uint4*)&Ah[r][cq * 8] = *(const uint4*)
                &hp16[(long)(row0 + r) * C_D + k0 + cq * 8];
        }
        #pragma unroll
        for (int j = 0; j < 6; ++j) {
            int r = tid >> 2, cq = tid & 3;
            int gate = j % 3;
            const _Float16* W = (j < 3) ? wih16 : whh16;
            *(uint4*)&Bt[j][r][cq * 8] = *(const uint4*)
                &W[(long)(gate * C_D + col0 + r) * C_D + k0 + cq * 8];
        }
#endif
        __syncthreads();

        union Frag { v16h v; v8h h[2]; };
        Frag ax, ah;
        ax.h[0] = *(const v8h*)&Ax[wave * 16 + m_][g * 16];
        ax.h[1] = *(const v8h*)&Ax[wave * 16 + m_][g * 16 + 8];
        ah.h[0] = *(const v8h*)&Ah[wave * 16 + m_][g * 16];
        ah.h[1] = *(const v8h*)&Ah[wave * 16 + m_][g * 16 + 8];

        #pragma unroll
        for (int s = 0; s < 2; ++s) {
            v16h b;
            b = *(const v16h*)&Bt[0][s * 16 + m_][g * 16];  // ih_r
            accR[s]  = wmma16(ax.v, b, accR[s]);
            b = *(const v16h*)&Bt[3][s * 16 + m_][g * 16];  // hh_r
            accR[s]  = wmma16(ah.v, b, accR[s]);
            b = *(const v16h*)&Bt[1][s * 16 + m_][g * 16];  // ih_z
            accZ[s]  = wmma16(ax.v, b, accZ[s]);
            b = *(const v16h*)&Bt[4][s * 16 + m_][g * 16];  // hh_z
            accZ[s]  = wmma16(ah.v, b, accZ[s]);
            b = *(const v16h*)&Bt[2][s * 16 + m_][g * 16];  // ih_n
            accNi[s] = wmma16(ax.v, b, accNi[s]);
            b = *(const v16h*)&Bt[5][s * 16 + m_][g * 16];  // hh_n
            accNh[s] = wmma16(ah.v, b, accNh[s]);
        }
        __syncthreads();
    }

    // gate math + blend
    #pragma unroll
    for (int s = 0; s < 2; ++s) {
        int col = col0 + s * 16 + m_;
        float br  = b_ih[col]       + b_hh[col];
        float bz  = b_ih[col + C_D] + b_hh[col + C_D];
        float bin = b_ih[col + 2 * C_D];
        float bhn = b_hh[col + 2 * C_D];
        #pragma unroll
        for (int r = 0; r < 8; ++r) {
            int row = row0 + wave * 16 + 8 * g + r;
            float rg = fast_sigmoid(accR[s][r] + br);
            float zg = fast_sigmoid(accZ[s][r] + bz);
            float ng = fast_tanh(accNi[s][r] + bin + rg * (accNh[s][r] + bhn));
            float hp = hp32[(long)row * C_D + col];
            float hn = (1.0f - zg) * ng + zg * hp;
            hn32[(long)row * C_D + col] = hn;
            hn16[(long)row * C_D + col] = (_Float16)hn;
        }
    }
}

// ---------------------------------------------------------------------------
// Host-side launcher
// ---------------------------------------------------------------------------
extern "C" void kernel_launch(void* const* d_in, const int* in_sizes, int n_in,
                              void* d_out, int out_size, void* d_ws, size_t ws_size,
                              hipStream_t stream) {
    const float* x    = (const float*)d_in[0];
    const float* c1w  = (const float*)d_in[1];
    const float* c1b  = (const float*)d_in[2];
    const float* c2w  = (const float*)d_in[3];
    const float* c2b  = (const float*)d_in[4];
    const float* wih  = (const float*)d_in[5];
    const float* whh  = (const float*)d_in[6];
    const float* bih  = (const float*)d_in[7];
    const float* bhh  = (const float*)d_in[8];
    const float* linw = (const float*)d_in[9];
    const float* linb = (const float*)d_in[10];

    // ---- workspace carve-out (256B aligned regions) ----
    char*  ws = (char*)d_ws;
    size_t o  = 0;
    auto carve = [&](size_t bytes) {
        char* p = ws + o;
        o += (bytes + 255) & ~(size_t)255;
        return p;
    };
    _Float16* c1w16 = (_Float16*)carve((size_t)C_C1 * C_A * 2);
    _Float16* c2w16 = (_Float16*)carve((size_t)C_D * C_C1 * 2);
    _Float16* wih16 = (_Float16*)carve((size_t)3 * C_D * C_D * 2);
    _Float16* whh16 = (_Float16*)carve((size_t)3 * C_D * C_D * 2);
    _Float16* lin16 = (_Float16*)carve((size_t)C_E * C_D * 2);
    _Float16* h1    = (_Float16*)carve((size_t)C_M * C_C1 * 2);
    _Float16* h2    = (_Float16*)carve((size_t)C_M * C_D * 2);
    float*    h32[2]; _Float16* h16[2];
    h32[0] = (float*)carve((size_t)C_B * C_D * 4);
    h32[1] = (float*)carve((size_t)C_B * C_D * 4);
    h16[0] = (_Float16*)carve((size_t)C_B * C_D * 2);
    h16[1] = (_Float16*)carve((size_t)C_B * C_D * 2);

    // ---- weight conversion f32 -> f16 ----
    auto cvt = [&](const float* s, _Float16* d, int n) {
        cvt_f32_to_f16_kernel<<<(n + 255) / 256, 256, 0, stream>>>(s, d, n);
    };
    cvt(c1w,  c1w16, C_C1 * C_A);
    cvt(c2w,  c2w16, C_D * C_C1);
    cvt(wih,  wih16, 3 * C_D * C_D);
    cvt(whh,  whh16, 3 * C_D * C_D);
    cvt(linw, lin16, C_E * C_D);

    // ---- h0 = 0 ----
    {
        int n = C_B * C_D;
        zero_h_kernel<<<(n + 255) / 256, 256, 0, stream>>>(h32[0], h16[0], n);
    }

    // ---- conv1: [26624 x 8192] x [8192 x 256] + bias, threshold -> f16 ----
    {
        dim3 grid(C_M / 128, C_C1 / 64);
        gemm_tile_kernel<false, 0><<<grid, 128, 0, stream>>>(
            x, C_A, c1w16, C_A, c1b, h1, C_C1, C_A);
    }
    // ---- conv2: [26624 x 256] x [256 x 512] + bias, threshold -> f16 ----
    {
        dim3 grid(C_M / 128, C_D / 64);
        gemm_tile_kernel<true, 0><<<grid, 128, 0, stream>>>(
            h1, C_C1, c2w16, C_C1, c2b, h2, C_D, C_C1);
    }
    // ---- GRU: 26 sequential steps, ping-pong h buffers ----
    {
        dim3 grid(C_B / 64, C_D / 32);
        for (int t = 0; t < C_L; ++t) {
            int src = t & 1, dst = (t + 1) & 1;
            gru_step_kernel<<<grid, 128, 0, stream>>>(
                h2, t, h16[src], h32[src], wih16, whh16, bih, bhh,
                h16[dst], h32[dst]);
        }
        // L=26 is even -> final hidden lands in buffer 0
    }
    // ---- final linear: [1024 x 512] x [512 x 1024] + bias -> f32 d_out ----
    {
        dim3 grid(C_B / 128, C_E / 64);
        gemm_tile_kernel<true, 1><<<grid, 128, 0, stream>>>(
            h16[0], C_D, lin16, C_D, linb, d_out, C_E, C_D);
    }
}
